// GAT_20383914787079
// MI455X (gfx1250) — matrix operand order repaired and checked by reading, not compiled, not run on hardware
//
#include <hip/hip_runtime.h>
#include <hip/hip_bf16.h>
#include <stdint.h>

typedef __attribute__((ext_vector_type(16))) __bf16 v16bf;
typedef __attribute__((ext_vector_type(8)))  float  v8f;
typedef __attribute__((ext_vector_type(4)))  int    v4i;

typedef __attribute__((address_space(1))) v4i as1_v4i;
typedef __attribute__((address_space(3))) v4i as3_v4i;

#define NEG_SLOPE 0.2f

// Async global->LDS staging path (gfx1250). Guarded so host-pass / older
// toolchains compile the plain-load fallback.
#if defined(__AMDGCN__) && __has_builtin(__builtin_amdgcn_global_load_async_to_lds_b128)
#define HAVE_ASYNC_LDS 1
#else
#define HAVE_ASYNC_LDS 0
#endif

// ---------------------------------------------------------------------------
// w_out[h*Fin + f] = sum_d W[h][f][d] * a[h][d]      (W: [2][Fin][D], a: [2][D])
// ---------------------------------------------------------------------------
__global__ void attn_vec_kernel(const float* __restrict__ W,
                                const float* __restrict__ a,
                                float* __restrict__ w_out,
                                int Fin, int D) {
    int idx = blockIdx.x * blockDim.x + threadIdx.x;
    if (idx >= 2 * Fin) return;
    int h = idx / Fin;
    int f = idx - h * Fin;
    const float* wr = W + ((size_t)h * Fin + f) * D;
    const float* ar = a + (size_t)h * D;
    float s = 0.f;
    for (int d = 0; d < D; ++d) s = fmaf(wr[d], ar[d], s);
    w_out[idx] = s;
}

// ---------------------------------------------------------------------------
// WT[b][c][r] = bf16(W[b][r][c])   (N-major bf16 weights for WMMA B-fragments)
// ---------------------------------------------------------------------------
__global__ void transpose_bf16_kernel(const float* __restrict__ W,
                                      __bf16* __restrict__ WT,
                                      int R, int C, int B) {
    int idx = blockIdx.x * blockDim.x + threadIdx.x;
    int total = B * R * C;
    if (idx >= total) return;
    int b  = idx / (R * C);
    int rc = idx - b * (R * C);
    int r  = rc / C;
    int c  = rc - r * C;
    WT[((size_t)b * C + c) * R + r] = (__bf16)W[idx];
}

// ---------------------------------------------------------------------------
// Fused attention + aggregation. One block per target row m.
//   e_self[h]  = x_self[m] . w_as[h]
//   e[k,h]     = leakyrelu(e_self[h] + x_neigh[m,k] . w_an[h])
//   alpha      = softmax_k(e)
//   xbar[m,h,:] = sum_k alpha[k,h] * x_neigh[m,k,:]
// blockDim.x == FIN (128 or 256). Streams x_neigh exactly once; the K x FIN
// neighbor tile is DMA'd to LDS with global_load_async_to_lds_b128 and the
// e_self reduction overlaps the async transfer.
// ---------------------------------------------------------------------------
template <int FIN, int KMAX>
__global__ void __launch_bounds__(FIN)
gat_aggregate_kernel(const float* __restrict__ x_self,
                     const float* __restrict__ x_neigh,
                     const float* __restrict__ w_as,   // [2][FIN]
                     const float* __restrict__ w_an,   // [2][FIN]
                     float* __restrict__ xbar,         // [M][2][FIN]
                     int K) {
    constexpr int NW = FIN / 32;
    __shared__ float s_xn[KMAX][FIN];
    __shared__ float s_part[KMAX + 1][2][NW];
    __shared__ float s_alpha[KMAX][2];

    const int f    = threadIdx.x;
    const int m    = blockIdx.x;
    const int lane = f & 31;
    const int wave = f >> 5;

    const float* xnb = x_neigh + (size_t)m * K * FIN;
    const int nfloat = K * FIN;

#if HAVE_ASYNC_LDS
    // Bulk-issue the whole neighbor tile as async b128 DMA into LDS.
    // Low 32 bits of a generic shared pointer == LDS byte offset.
    {
        float* s_flat = &s_xn[0][0];
        for (int i = 4 * f; i < nfloat; i += 4 * FIN) {
            as1_v4i* gp = (as1_v4i*)(uintptr_t)(const void*)(xnb + i);
            as3_v4i* lp = (as3_v4i*)(uint32_t)(uintptr_t)(void*)(s_flat + i);
            __builtin_amdgcn_global_load_async_to_lds_b128(gp, lp, 0, 0);
        }
    }
#else
    for (int k = 0; k < K; ++k) s_xn[k][f] = xnb[(size_t)k * FIN + f];
#endif

    // --- e_self partials (overlaps the async DMA) ---
    const float was0 = w_as[f], was1 = w_as[FIN + f];
    const float wan0 = w_an[f], wan1 = w_an[FIN + f];
    float xs = x_self[(size_t)m * FIN + f];
    float p0 = xs * was0, p1 = xs * was1;
    #pragma unroll
    for (int off = 16; off; off >>= 1) {
        p0 += __shfl_xor(p0, off, 32);
        p1 += __shfl_xor(p1, off, 32);
    }
    if (lane == 0) { s_part[KMAX][0][wave] = p0; s_part[KMAX][1][wave] = p1; }

#if HAVE_ASYNC_LDS
#if __has_builtin(__builtin_amdgcn_s_wait_asynccnt)
    __builtin_amdgcn_s_wait_asynccnt(0);
#else
    asm volatile("s_wait_asynccnt 0" ::: "memory");
#endif
#endif
    __syncthreads();

    // --- neighbor logit partials from the LDS tile ---
    for (int k = 0; k < K; ++k) {
        float v = s_xn[k][f];
        float q0 = v * wan0, q1 = v * wan1;
        #pragma unroll
        for (int off = 16; off; off >>= 1) {
            q0 += __shfl_xor(q0, off, 32);
            q1 += __shfl_xor(q1, off, 32);
        }
        if (lane == 0) { s_part[k][0][wave] = q0; s_part[k][1][wave] = q1; }
    }
    __syncthreads();

    // --- softmax over K per head (2 threads) ---
    if (f < 2) {
        const int h = f;
        float es = 0.f;
        for (int w = 0; w < NW; ++w) es += s_part[KMAX][h][w];
        float e[KMAX];
        float mx = -1e30f;
        for (int k = 0; k < K; ++k) {
            float en = 0.f;
            for (int w = 0; w < NW; ++w) en += s_part[k][h][w];
            float t = es + en;
            t = (t > 0.f) ? t : NEG_SLOPE * t;
            e[k] = t;
            mx = fmaxf(mx, t);
        }
        float s = 0.f;
        for (int k = 0; k < K; ++k) { float ex = __expf(e[k] - mx); e[k] = ex; s += ex; }
        float inv = 1.0f / s;
        for (int k = 0; k < K; ++k) s_alpha[k][h] = e[k] * inv;
    }
    __syncthreads();

    // --- alpha-weighted aggregation ---
    float acc0 = 0.f, acc1 = 0.f;
    for (int k = 0; k < K; ++k) {
        float v = s_xn[k][f];
        acc0 = fmaf(s_alpha[k][0], v, acc0);
        acc1 = fmaf(s_alpha[k][1], v, acc1);
    }
    xbar[((size_t)m * 2 + 0) * FIN + f] = acc0;
    xbar[((size_t)m * 2 + 1) * FIN + f] = acc1;
}

// ---------------------------------------------------------------------------
// Per-head GEMM:  C[m, h*coff + n] = sum_k A[m*lda + h*aoff + k] * W_h[k][n]
// A: f32 (converted to bf16 in-register), BT: bf16 N-major ([n][k] contiguous),
// C: f32.  One wave computes a 16x64 tile via 4x v_wmma_f32_16x16x32_bf16.
// grid = (M/64, N/64, H), block = 128 threads (4 waves).
// ---------------------------------------------------------------------------
__global__ void __launch_bounds__(128)
gemm_wmma_kernel(const float* __restrict__ A, int lda, int aoff,
                 const __bf16* __restrict__ BT, int boff,
                 float* __restrict__ C, int ldc, int coff,
                 int M, int K) {
    const int lane    = threadIdx.x & 31;
    const int wave    = threadIdx.x >> 5;
    const int h       = blockIdx.z;
    const int rowbase = (blockIdx.x * 4 + wave) * 16;
    const int colbase = blockIdx.y * 64;
    if (rowbase >= M) return;

    const float*  Ar = A  + (size_t)(rowbase + (lane & 15)) * lda + (size_t)h * aoff;
    const __bf16* Bt = BT + (size_t)h * boff;

    v8f acc[4] = {};

    // A fragment K-lane mapping (contiguous, matches B):
    //   lanes 0-15 : K = 0..15;  lanes 16-31 : K = 16..31
    const int ak0 = (lane < 16) ? 0 : 16;
    const int ak1 = (lane < 16) ? 8 : 24;
    // B fragment: lanes 0-15 hold K = 0..15, lanes 16-31 hold K = 16..31
    const int bk  = (lane < 16) ? 0 : 16;

    for (int kb = 0; kb < K; kb += 32) {
        v16bf a;
        #pragma unroll
        for (int i = 0; i < 8; ++i) a[i]     = (__bf16)Ar[kb + ak0 + i];
        #pragma unroll
        for (int i = 0; i < 8; ++i) a[8 + i] = (__bf16)Ar[kb + ak1 + i];

        #pragma unroll
        for (int ct = 0; ct < 4; ++ct) {
            const int n = colbase + ct * 16 + (lane & 15);
            const __bf16* bp = Bt + (size_t)n * K + kb + bk;
            v16bf b;
            #pragma unroll
            for (int i = 0; i < 16; ++i) b[i] = bp[i];
            acc[ct] = __builtin_amdgcn_wmma_f32_16x16x32_bf16(
                false, a, false, b, (short)0, acc[ct], false, false);
        }
    }

    // C/D layout: VGPR j -> row rowbase + j (+8 for lanes 16-31), col = lane&15
    const int rw = rowbase + ((lane < 16) ? 0 : 8);
    const int cn = colbase + (lane & 15);
    #pragma unroll
    for (int ct = 0; ct < 4; ++ct) {
        #pragma unroll
        for (int j = 0; j < 8; ++j) {
            C[(size_t)(rw + j) * ldc + (size_t)h * coff + cn + ct * 16] = acc[ct][j];
        }
    }
}

// ---------------------------------------------------------------------------
extern "C" void kernel_launch(void* const* d_in, const int* in_sizes, int n_in,
                              void* d_out, int out_size, void* d_ws, size_t ws_size,
                              hipStream_t stream) {
    const float* h0  = (const float*)d_in[0];   // [1024,128]
    const float* h1  = (const float*)d_in[1];   // [10240,128]
    const float* h2  = (const float*)d_in[2];   // [256000,128]
    const float* W0  = (const float*)d_in[3];   // [2,128,128]
    const float* as0 = (const float*)d_in[4];   // [2,128]
    const float* an0 = (const float*)d_in[5];   // [2,128]
    const float* W1  = (const float*)d_in[6];   // [2,256,128]
    const float* as1 = (const float*)d_in[7];   // [2,128]
    const float* an1 = (const float*)d_in[8];   // [2,128]
    const float* fcW = (const float*)d_in[9];   // [256,128]
    float* out = (float*)d_out;                 // [1024,128]

    const int B = 1024, F = 128, D = 128;
    const int M1 = 10240;                        // B * r0
    (void)in_sizes; (void)n_in; (void)out_size; (void)ws_size;

    // ---- workspace carve-up (256B aligned) ----
    char* ws = (char*)d_ws;
    size_t off = 0;
    auto carve = [&](size_t bytes) {
        char* p = ws + off;
        off += (bytes + 255) & ~(size_t)255;
        return p;
    };
    float*  w_as0   = (float*) carve(2 * 128 * 4);
    float*  w_an0   = (float*) carve(2 * 128 * 4);
    float*  w_as1   = (float*) carve(2 * 256 * 4);
    float*  w_an1   = (float*) carve(2 * 256 * 4);
    __bf16* W0T     = (__bf16*)carve((size_t)2 * 128 * 128 * 2);  // [h][d=128][f=128]
    __bf16* W1T     = (__bf16*)carve((size_t)2 * 128 * 256 * 2);  // [h][d=128][f=256]
    __bf16* fcWT    = (__bf16*)carve((size_t)128 * 256 * 2);      // [o=128][f=256]
    float*  xbar1   = (float*) carve((size_t)M1 * 2 * 128 * 4);   // [10240][2][128]
    float*  hidden1 = (float*) carve((size_t)M1 * 256 * 4);       // [10240][256]
    float*  xbar0   = (float*) carve((size_t)B * 2 * 128 * 4);    // [1024][2][128]
    float*  hidden0 = (float*) carve((size_t)B * 256 * 4);        // [1024][256]
    float*  xbar0b  = (float*) carve((size_t)B * 2 * 256 * 4);    // [1024][2][256]
    float*  hid0b   = (float*) carve((size_t)B * 256 * 4);        // [1024][256]

    // ---- 1. precompute attention vectors  w = W_h a_h ----
    attn_vec_kernel<<<1, 256, 0, stream>>>(W0, as0, w_as0, F, D);
    attn_vec_kernel<<<1, 256, 0, stream>>>(W0, an0, w_an0, F, D);
    attn_vec_kernel<<<2, 256, 0, stream>>>(W1, as1, w_as1, 2 * D, D);
    attn_vec_kernel<<<2, 256, 0, stream>>>(W1, an1, w_an1, 2 * D, D);

    // ---- 2. pre-transpose weights into N-major bf16 ----
    transpose_bf16_kernel<<<(2 * 128 * 128 + 255) / 256, 256, 0, stream>>>(W0, W0T, 128, 128, 2);
    transpose_bf16_kernel<<<(2 * 256 * 128 + 255) / 256, 256, 0, stream>>>(W1, W1T, 256, 128, 2);
    transpose_bf16_kernel<<<(256 * 128 + 255) / 256, 256, 0, stream>>>(fcW, fcWT, 256, 128, 1);

    // ---- 3. layer 0, j=1: aggregate h2 under (h1 self), project -> hidden1 ----
    gat_aggregate_kernel<128, 25><<<M1, 128, 0, stream>>>(h1, h2, w_as0, w_an0, xbar1, 25);
    gemm_wmma_kernel<<<dim3(M1 / 64, 2, 2), 128, 0, stream>>>(
        xbar1, 256, 128, W0T, 128 * 128, hidden1, 256, 128, M1, 128);

    // ---- 4. layer 0, j=0: aggregate h1 under (h0 self), project -> hidden0 ----
    gat_aggregate_kernel<128, 10><<<B, 128, 0, stream>>>(h0, h1, w_as0, w_an0, xbar0, 10);
    gemm_wmma_kernel<<<dim3(B / 64, 2, 2), 128, 0, stream>>>(
        xbar0, 256, 128, W0T, 128 * 128, hidden0, 256, 128, B, 128);

    // ---- 5. layer 1, j=0: aggregate hidden1 under (hidden0 self), project ----
    gat_aggregate_kernel<256, 10><<<B, 256, 0, stream>>>(hidden0, hidden1, w_as1, w_an1, xbar0b, 10);
    gemm_wmma_kernel<<<dim3(B / 64, 2, 2), 128, 0, stream>>>(
        xbar0b, 512, 256, W1T, 128 * 256, hid0b, 256, 128, B, 256);

    // ---- 6. final FC: out = hid0b @ fc_W ----
    gemm_wmma_kernel<<<dim3(B / 64, 2, 1), 128, 0, stream>>>(
        hid0b, 256, 0, fcWT, 0, out, 128, 0, B, 256);
}
